// Transformer_78718160601474
// MI455X (gfx1250) — matrix operand order repaired and checked by reading, not compiled, not run on hardware
//
#include <hip/hip_runtime.h>
#include <math.h>

// ---------------------------------------------------------------------------
// Types for CDNA5 WMMA (gfx1250, wave32)
// ---------------------------------------------------------------------------
typedef __attribute__((ext_vector_type(16))) __bf16 v16bf;
typedef __attribute__((ext_vector_type(8)))  float  v8f;
typedef __attribute__((ext_vector_type(4)))  int    v4i;

#if defined(__AMDGCN__) && __has_builtin(__builtin_amdgcn_global_load_async_to_lds_b128)
#define HAVE_ASYNC 1
#else
#define HAVE_ASYNC 0
#endif

__device__ __forceinline__ v8f wmma_bf16(v16bf a, v16bf b, v8f c) {
  return __builtin_amdgcn_wmma_f32_16x16x32_bf16(
      /*neg_a=*/false, a, /*neg_b=*/false, b,
      /*c_mod=*/(short)0, c, /*reuse_a=*/false, /*reuse_b=*/false);
}

__device__ __forceinline__ unsigned short f2bf(float f) {
  unsigned u = __float_as_uint(f);
  u += 0x7fffu + ((u >> 16) & 1u);     // round-to-nearest-even
  return (unsigned short)(u >> 16);
}

// Copy 16B global -> LDS. Async (ASYNCcnt) path on CDNA5, else ld+st fallback.
__device__ __forceinline__ void cp16(unsigned short* dst, const unsigned short* src) {
#if HAVE_ASYNC
  __builtin_amdgcn_global_load_async_to_lds_b128((v4i*)src, (v4i*)dst, 0, 0);
#else
  __builtin_prefetch(src + 32, 0, 1);   // next K-panel -> global_prefetch_b8
  *(uint4*)dst = *(const uint4*)src;
#endif
}

__device__ __forceinline__ void wait_async() {
#if HAVE_ASYNC
# if __has_builtin(__builtin_amdgcn_s_wait_asynccnt)
  __builtin_amdgcn_s_wait_asynccnt(0);
# else
  asm volatile("s_wait_asynccnt 0x0" ::: "memory");
# endif
#endif
}

// A-fragment (16x32 bf16): lane's 16 K values contiguous (lane half h -> K{kb+16h..kb+16h+15}), 16B x2
__device__ __forceinline__ v16bf ld_fragA(const unsigned short* p) {
  union { v16bf v; uint4 q[2]; } u;
  u.q[0] = *(const uint4*)(p);
  u.q[1] = *(const uint4*)(p + 8);
  return u.v;
}
// B-fragment (32x16 bf16): lane's 16 K values contiguous (32B)
__device__ __forceinline__ v16bf ld_fragB(const unsigned short* p) {
  union { v16bf v; uint4 q[2]; } u;
  u.q[0] = *(const uint4*)(p);
  u.q[1] = *(const uint4*)(p + 8);
  return u.v;
}

// ---------------------------------------------------------------------------
// Batched bf16 GEMM: C[z] = A[z](MxK) * B[z](NxK, K-contiguous) [+bias][ReLU]
// A,B bf16 in global; 128x64x32 block tile; 8 waves, 32x32 per wave;
// double-buffered LDS staged with async-load-to-LDS; f32 accumulate.
// Output fp32 or bf16. Requires M%128==0, N%64==0, K%32==0, ld*%8==0.
// ---------------------------------------------------------------------------
#define TM 128
#define TN 64
#define TK 32

__global__ void __launch_bounds__(256) k_gemm_bf16(
    const unsigned short* __restrict__ A,
    const unsigned short* __restrict__ B,
    void* __restrict__ C,
    const float* __restrict__ bias,
    int K, int lda, int ldb, int ldc,
    long long sA, long long sB, long long sC,
    int outBf16, int do_relu)
{
  __shared__ __align__(16) unsigned short As[2][TM * TK];
  __shared__ __align__(16) unsigned short Bs[2][TN * TK];

  const long long z = blockIdx.z;
  const unsigned short* Ab = A + sA * z;
  const unsigned short* Bb = B + sB * z;

  const int m0  = blockIdx.y * TM;
  const int n0  = blockIdx.x * TN;
  const int tid = threadIdx.x;
  const int lane = tid & 31;
  const int wv   = tid >> 5;
  const int rowBase = (wv >> 1) * 32;
  const int colBase = (wv & 1) * 32;
  const int hlf = lane >> 4;
  const int l15 = lane & 15;

  // staging map: thread -> (row 0..63 [+64], 8-elem chunk)
  const int ra = tid >> 2;
  const int c8 = (tid & 3) << 3;

  v8f c00 = {0.f,0.f,0.f,0.f,0.f,0.f,0.f,0.f};
  v8f c01 = c00, c10 = c00, c11 = c00;

  const int nT = K / TK;

  auto stage = [&](int bi, int k0) {
    #pragma unroll
    for (int p = 0; p < 2; ++p) {
      const int row = ra + (p << 6);
      cp16(&As[bi][row * TK + c8], Ab + (long long)(m0 + row) * lda + (k0 + c8));
    }
    cp16(&Bs[bi][ra * TK + c8], Bb + (long long)(n0 + ra) * ldb + (k0 + c8));
  };

  stage(0, 0);
  for (int t = 0; t < nT; ++t) {
    const int cur = t & 1;
    wait_async();            // my async tile landed in LDS
    __syncthreads();         // everyone's landed
    if (t + 1 < nT) stage(cur ^ 1, (t + 1) * TK);   // prefetch next tile

    const unsigned short* As_ = As[cur];
    const unsigned short* Bs_ = Bs[cur];
    const v16bf a0 = ld_fragA(&As_[(rowBase +  0 + l15) * TK + (hlf << 4)]);
    const v16bf a1 = ld_fragA(&As_[(rowBase + 16 + l15) * TK + (hlf << 4)]);
    const v16bf b0 = ld_fragB(&Bs_[(colBase +  0 + l15) * TK + (hlf << 4)]);
    const v16bf b1 = ld_fragB(&Bs_[(colBase + 16 + l15) * TK + (hlf << 4)]);

    c00 = wmma_bf16(a0, b0, c00);
    c01 = wmma_bf16(a0, b1, c01);
    c10 = wmma_bf16(a1, b0, c10);
    c11 = wmma_bf16(a1, b1, c11);
    __syncthreads();         // done reading buf[cur] before t+1 overwrites it
  }

  float*          Cf = (float*)C          + sC * z;
  unsigned short* Ch = (unsigned short*)C + sC * z;
  auto emit = [&](const v8f& acc, int tr, int tc) {
    const int r   = m0 + rowBase + tr + (hlf << 3);
    const int col = n0 + colBase + tc + l15;
    const float bv = bias ? bias[col] : 0.f;
    #pragma unroll
    for (int j = 0; j < 8; ++j) {
      float v = acc[j] + bv;
      if (do_relu) v = fmaxf(v, 0.f);
      if (outBf16) Ch[(long long)(r + j) * ldc + col] = f2bf(v);
      else         Cf[(long long)(r + j) * ldc + col] = v;
    }
  };
  emit(c00,  0,  0);
  emit(c01,  0, 16);
  emit(c10, 16,  0);
  emit(c11, 16, 16);
}

// ---------------------------------------------------------------------------
// Weight cast+transpose: fp32 [z][R][C] -> bf16 [z][C][R] (K-contiguous B)
// ---------------------------------------------------------------------------
__global__ void __launch_bounds__(256) k_castT(const float* __restrict__ in,
                                               unsigned short* __restrict__ out,
                                               int R, int Cc)
{
  const long long base = (long long)blockIdx.z * R * Cc;
  const long long f = (long long)blockIdx.x * 256 + threadIdx.x;
  const int r = (int)(f / Cc);
  const int c = (int)(f % Cc);
  out[base + (long long)c * R + r] = f2bf(in[base + f]);
}

// bf16 [z][R][C] -> bf16 [z][C][R]  (used to transpose V)
__global__ void __launch_bounds__(256) k_transH(const unsigned short* __restrict__ in,
                                                unsigned short* __restrict__ out,
                                                int R, int Cc)
{
  const long long base = (long long)blockIdx.z * R * Cc;
  const long long f = (long long)blockIdx.x * 256 + threadIdx.x;
  const int r = (int)(f / Cc);
  const int c = (int)(f % Cc);
  out[base + (long long)c * R + r] = in[base + f];
}

// ---------------------------------------------------------------------------
// Embedding * sqrt(E) + sinusoidal PE (E=512); emits fp32 + bf16
// ---------------------------------------------------------------------------
__global__ void __launch_bounds__(256) k_embed(const int* __restrict__ ids,
                                               const float* __restrict__ emb,
                                               float* __restrict__ out32,
                                               unsigned short* __restrict__ out16,
                                               float scale)
{
  const long long idx = (long long)blockIdx.x * 256 + threadIdx.x;
  const int e = (int)(idx & 511);
  const long long bs = idx >> 9;
  const int s = (int)(bs & 511);
  const int tok = ids[bs];
  const float LN1E4_O256 = 9.210340371976184f / 256.0f;   // ln(10000)/256
  float pe;
  if (e < 256) pe = __sinf((float)s * __expf(-(float)e * LN1E4_O256));
  else         pe = __cosf((float)s * __expf(-(float)(e - 256) * LN1E4_O256));
  const float v = emb[(long long)tok * 512 + e] * scale + pe;
  out32[idx] = v;
  out16[idx] = f2bf(v);
}

// ---------------------------------------------------------------------------
// Row softmax over S=512 with 1/sqrt(Sk) scale + optional causal mask.
// fp32 in-place, plus bf16 copy for the following WMMA GEMM.
// ---------------------------------------------------------------------------
__global__ void __launch_bounds__(256) k_softmax_rows(float* __restrict__ logits,
                                                      unsigned short* __restrict__ p16,
                                                      int causal, float scale)
{
  __shared__ float red[256];
  const long long row = blockIdx.x;
  const int q = (int)(row & 511);
  float* p = logits + (row << 9);
  const int t = threadIdx.x;

  float v0 = p[t] * scale;
  float v1 = p[t + 256] * scale;
  if (causal) {
    if (t > q)       v0 = -1e30f;
    if (t + 256 > q) v1 = -1e30f;
  }
  red[t] = fmaxf(v0, v1);
  __syncthreads();
  for (int off = 128; off > 0; off >>= 1) { if (t < off) red[t] = fmaxf(red[t], red[t + off]); __syncthreads(); }
  const float m = red[0];
  __syncthreads();
  const float e0 = __expf(v0 - m);
  const float e1 = __expf(v1 - m);
  red[t] = e0 + e1;
  __syncthreads();
  for (int off = 128; off > 0; off >>= 1) { if (t < off) red[t] += red[t + off]; __syncthreads(); }
  const float inv = 1.0f / red[0];
  const float r0 = e0 * inv, r1 = e1 * inv;
  p[t]       = r0;        p[t + 256]       = r1;
  p16[(row << 9) + t] = f2bf(r0);
  p16[(row << 9) + t + 256] = f2bf(r1);
}

// ---------------------------------------------------------------------------
// out = [ReLU](LayerNorm(x + a) * g + b), width 512, eps 1e-3; fp32 + bf16
// ---------------------------------------------------------------------------
__global__ void __launch_bounds__(256) k_add_ln(const float* __restrict__ x,
                                                const float* __restrict__ a,
                                                const float* __restrict__ g,
                                                const float* __restrict__ bb,
                                                float* __restrict__ out32,
                                                unsigned short* __restrict__ out16,
                                                int do_relu)
{
  __shared__ float red[256];
  const long long row = blockIdx.x;
  const float* xr = x + (row << 9);
  const float* ar = a + (row << 9);
  const int t = threadIdx.x;

  const float v0 = xr[t] + ar[t];
  const float v1 = xr[t + 256] + ar[t + 256];
  red[t] = v0 + v1;
  __syncthreads();
  for (int off = 128; off > 0; off >>= 1) { if (t < off) red[t] += red[t + off]; __syncthreads(); }
  const float mean = red[0] * (1.0f / 512.0f);
  __syncthreads();
  const float d0 = v0 - mean, d1 = v1 - mean;
  red[t] = d0 * d0 + d1 * d1;
  __syncthreads();
  for (int off = 128; off > 0; off >>= 1) { if (t < off) red[t] += red[t + off]; __syncthreads(); }
  const float rs = rsqrtf(red[0] * (1.0f / 512.0f) + 1e-3f);
  float o0 = d0 * rs * g[t] + bb[t];
  float o1 = d1 * rs * g[t + 256] + bb[t + 256];
  if (do_relu) { o0 = fmaxf(o0, 0.f); o1 = fmaxf(o1, 0.f); }
  out32[(row << 9) + t]       = o0;
  out32[(row << 9) + t + 256] = o1;
  out16[(row << 9) + t]       = f2bf(o0);
  out16[(row << 9) + t + 256] = f2bf(o1);
}

// ---------------------------------------------------------------------------
// Concat heads (bf16): in[H=8][B=8][S=512][D=64] -> out[B][S][512]
// ---------------------------------------------------------------------------
__global__ void __launch_bounds__(256) k_concat_heads(const unsigned short* __restrict__ in,
                                                      unsigned short* __restrict__ out)
{
  const long long idx = (long long)blockIdx.x * 256 + threadIdx.x;
  const int e = (int)(idx & 511);
  const long long bs = idx >> 9;
  const int s = (int)(bs & 511);
  const int b = (int)(bs >> 9);
  const int h = e >> 6;
  const int d = e & 63;
  out[idx] = in[(((long long)(h * 8 + b) * 512 + s) << 6) + d];
}

// ---------------------------------------------------------------------------
// Host orchestration
// ---------------------------------------------------------------------------
extern "C" void kernel_launch(void* const* d_in, const int* in_sizes, int n_in,
                              void* d_out, int out_size, void* d_ws, size_t ws_size,
                              hipStream_t stream) {
  (void)in_sizes; (void)n_in; (void)out_size; (void)ws_size;

  const int*   src_ids = (const int*)d_in[0];
  const int*   tgt_ids = (const int*)d_in[1];
  const float* src_emb = (const float*)d_in[2];
  const float* tgt_emb = (const float*)d_in[3];
  const float* enc_wk  = (const float*)d_in[4];
  const float* enc_wv  = (const float*)d_in[5];
  const float* enc_wq  = (const float*)d_in[6];
  const float* enc_dw  = (const float*)d_in[7];
  const float* enc_db  = (const float*)d_in[8];
  const float* enc_ffw = (const float*)d_in[9];
  const float* enc_ffb = (const float*)d_in[10];
  const float* enc_ln_g = (const float*)d_in[11];
  const float* enc_ln_b = (const float*)d_in[12];
  const float* dec_swk = (const float*)d_in[13];
  const float* dec_swv = (const float*)d_in[14];
  const float* dec_swq = (const float*)d_in[15];
  const float* dec_sdw = (const float*)d_in[16];
  const float* dec_sdb = (const float*)d_in[17];
  const float* dec_cwk = (const float*)d_in[18];
  const float* dec_cwv = (const float*)d_in[19];
  const float* dec_cwq = (const float*)d_in[20];
  const float* dec_cdw = (const float*)d_in[21];
  const float* dec_cdb = (const float*)d_in[22];
  const float* dec_ffw = (const float*)d_in[23];
  const float* dec_ffb = (const float*)d_in[24];
  const float* dec_ln_g = (const float*)d_in[25];
  const float* dec_ln_b = (const float*)d_in[26];
  const float* cls_w1  = (const float*)d_in[27];
  const float* cls_b1  = (const float*)d_in[28];
  const float* cls_w2  = (const float*)d_in[29];
  const float* cls_b2  = (const float*)d_in[30];

  const long long NE = 2097152;            // 8*512*512
  char* arena = (char*)d_ws;
  size_t off = 0;
  auto takeF = [&](long long n) { float* p = (float*)(arena + off);
                                  off = (off + (size_t)n * 4 + 15) & ~(size_t)15; return p; };
  auto takeH = [&](long long n) { unsigned short* p = (unsigned short*)(arena + off);
                                  off = (off + (size_t)n * 2 + 15) & ~(size_t)15; return p; };
  // fp32 (residual / softmax precision)
  float* X32     = takeF(NE);
  float* Y32     = takeF(NE);
  float* tmpA    = takeF(NE);              // GEMM fp32 outputs feeding LN
  float* hE32    = takeF(NE);
  float* encO32  = takeF(NE);
  float* h1_32   = takeF(NE);
  float* h2_32   = takeF(NE);
  float* dO32    = takeF(NE);
  float* logits  = takeF(16777216LL);      // [H*B][512][512]
  // bf16 (GEMM operands)
  unsigned short* X16    = takeH(NE);
  unsigned short* Y16    = takeH(NE);
  unsigned short* hE16   = takeH(NE);
  unsigned short* encO16 = takeH(NE);
  unsigned short* h1_16  = takeH(NE);
  unsigned short* h2_16  = takeH(NE);
  unsigned short* dO16   = takeH(NE);
  unsigned short* Q16    = takeH(NE);      // [H][4096][64]
  unsigned short* K16    = takeH(NE);      // [H][4096][64]  (already K-contig as B^T)
  unsigned short* V16    = takeH(NE);
  unsigned short* Vt16   = takeH(NE);      // [H*B][64][512]
  unsigned short* attp16 = takeH(NE);      // [H][B][S][64]
  unsigned short* atto16 = takeH(NE);      // [B][S][512]
  unsigned short* P16    = takeH(16777216LL);
  unsigned short* hid16  = takeH(65536000LL);
  // bf16 transposed weights
  unsigned short* wkT  = takeH(262144);    // [H][64][512]
  unsigned short* wvT  = takeH(262144);
  unsigned short* wqT  = takeH(262144);
  unsigned short* dwT  = takeH(262144);    // [512][512]
  unsigned short* ffwT = takeH(262144);
  unsigned short* w1T  = takeH(8192000LL); // [16000][512]
  unsigned short* w2T  = takeH(128000000LL); // [8000][16000]

  auto gemm = [&](const unsigned short* A, const unsigned short* B, void* Cm,
                  const float* bias, int M, int N, int Kd,
                  int lda, int ldb, int ldc,
                  long long sA, long long sB, long long sC,
                  int nb, int outBf16, int relu) {
    dim3 grid((unsigned)(N / TN), (unsigned)(M / TM), (unsigned)nb);
    k_gemm_bf16<<<grid, 256, 0, stream>>>(A, B, Cm, bias, Kd, lda, ldb, ldc,
                                          sA, sB, sC, outBf16, relu);
  };
  auto castT = [&](const float* in, unsigned short* out, int R, int Cc, int nb) {
    k_castT<<<dim3((unsigned)(((long long)R * Cc) / 256), 1, (unsigned)nb),
              256, 0, stream>>>(in, out, R, Cc);
  };

  const float inv_sqrt_sk = 0.04419417382415922f;   // 1/sqrt(512)
  const float emb_scale   = 22.627416997969522f;    // sqrt(512)

  // z enumerates h*8+b for attention GEMMs; all strides linear in z
  auto mha = [&](const unsigned short* xk, const unsigned short* xv,
                 const unsigned short* xq,
                 const float* wk, const float* wv, const float* wq,
                 const float* dw, const float* db, float* out32, int causal) {
    castT(wk, wkT, 512, 64, 8);
    castT(wv, wvT, 512, 64, 8);
    castT(wq, wqT, 512, 64, 8);
    castT(dw, dwT, 512, 512, 1);
    // per-head projections: [4096,512] x [512,64] -> bf16 [H][4096][64]
    gemm(xk, wkT, K16, nullptr, 4096, 64, 512, 512, 512, 64, 0, 32768, 262144, 8, 1, 0);
    gemm(xv, wvT, V16, nullptr, 4096, 64, 512, 512, 512, 64, 0, 32768, 262144, 8, 1, 0);
    gemm(xq, wqT, Q16, nullptr, 4096, 64, 512, 512, 512, 64, 0, 32768, 262144, 8, 1, 0);
    // V -> V^T per (h,b): [512][64] -> [64][512]
    k_transH<<<dim3(128, 1, 64), 256, 0, stream>>>(V16, Vt16, 512, 64);
    // scores: Q[512,64] x K^T -> fp32 [512,512] per (h,b)
    gemm(Q16, K16, logits, nullptr, 512, 512, 64, 64, 64, 512,
         32768, 32768, 262144, 64, 0, 0);
    k_softmax_rows<<<dim3(32768), dim3(256), 0, stream>>>(logits, P16, causal, inv_sqrt_sk);
    // att = P[512,512] x V[512,64]  (B = Vt, K-contiguous)
    gemm(P16, Vt16, attp16, nullptr, 512, 64, 512, 512, 512, 64,
         262144, 32768, 32768, 64, 1, 0);
    k_concat_heads<<<dim3(8192), dim3(256), 0, stream>>>(attp16, atto16);
    // output projection + bias -> fp32 (for LayerNorm residual path)
    gemm(atto16, dwT, out32, db, 4096, 512, 512, 512, 512, 512, 0, 0, 0, 1, 0, 0);
  };

  // embeddings + positional encoding
  k_embed<<<dim3(8192), dim3(256), 0, stream>>>(src_ids, src_emb, X32, X16, emb_scale);
  k_embed<<<dim3(8192), dim3(256), 0, stream>>>(tgt_ids, tgt_emb, Y32, Y16, emb_scale);

  // encoder (self-attn causally masked, per reference)
  mha(X16, X16, X16, enc_wk, enc_wv, enc_wq, enc_dw, enc_db, tmpA, 1);
  k_add_ln<<<dim3(4096), dim3(256), 0, stream>>>(X32, tmpA, enc_ln_g, enc_ln_b, hE32, hE16, 0);
  castT(enc_ffw, ffwT, 512, 512, 1);
  gemm(hE16, ffwT, tmpA, enc_ffb, 4096, 512, 512, 512, 512, 512, 0, 0, 0, 1, 0, 0);
  k_add_ln<<<dim3(4096), dim3(256), 0, stream>>>(hE32, tmpA, enc_ln_g, enc_ln_b, encO32, encO16, 0);

  // decoder: self-attn (causal), cross-attn, FF
  mha(Y16, Y16, Y16, dec_swk, dec_swv, dec_swq, dec_sdw, dec_sdb, tmpA, 1);
  k_add_ln<<<dim3(4096), dim3(256), 0, stream>>>(Y32, tmpA, dec_ln_g, dec_ln_b, h1_32, h1_16, 0);
  mha(encO16, encO16, h1_16, dec_cwk, dec_cwv, dec_cwq, dec_cdw, dec_cdb, tmpA, 0);
  k_add_ln<<<dim3(4096), dim3(256), 0, stream>>>(h1_32, tmpA, dec_ln_g, dec_ln_b, h2_32, h2_16, 0);
  castT(dec_ffw, ffwT, 512, 512, 1);
  gemm(h2_16, ffwT, tmpA, dec_ffb, 4096, 512, 512, 512, 512, 512, 0, 0, 0, 1, 0, 0);
  k_add_ln<<<dim3(4096), dim3(256), 0, stream>>>(h2_32, tmpA, dec_ln_g, dec_ln_b, dO32, dO16, 1);

  // classifier
  castT(cls_w1, w1T, 512, 16000, 1);
  gemm(dO16, w1T, hid16, cls_b1, 4096, 16000, 512, 512, 512, 16000, 0, 0, 0, 1, 1, 1);
  castT(cls_w2, w2T, 16000, 8000, 1);
  gemm(hid16, w2T, d_out, cls_b2, 4096, 8000, 16000, 16000, 16000, 8000, 0, 0, 0, 1, 0, 0);
}
